// CompressedLinear_52441550684585
// MI455X (gfx1250) — compile-verified, matrix operand order checked by reading, not run on hardware
//
#include <hip/hip_runtime.h>

// GEMM: out[t, r] = sum_k x[t,k] * (w_int8[r,k] * SCALE) + bias[r]
//   M = 128 tokens (4*32), N = 11008 rows, K = 4096.
// Memory-bound on streaming W (int32 in memory, 180 MB ~= 7.7us at 23.3 TB/s).
// WMMA f16 with f32 accumulation (int8 weights exact in f16); x staged to LDS
// via CDNA5 async-to-LDS loads (ASYNCcnt).

#define NROWS 11008
#define KCOLS 4096
#define MTOK  128
#define WSCALE 0.01f

typedef _Float16 v16h __attribute__((ext_vector_type(16)));
typedef _Float16 v8h  __attribute__((ext_vector_type(8)));
typedef _Float16 v4h  __attribute__((ext_vector_type(4)));
typedef float    v8f  __attribute__((ext_vector_type(8)));
typedef float    v4f  __attribute__((ext_vector_type(4)));
typedef int      v4i  __attribute__((ext_vector_type(4)));

// 16-byte payload type for the async-to-LDS builtin (per hipcc diagnostic:
// parameter is `int __vector(4) *`).
typedef int b128i __attribute__((vector_size(16)));

#if __has_builtin(__builtin_amdgcn_global_load_async_to_lds_b128)
#define HAS_ASYNC_LDS 1
#else
#define HAS_ASYNC_LDS 0
#endif

__device__ __forceinline__ void wait_async_zero() {
#if HAS_ASYNC_LDS
#if __has_builtin(__builtin_amdgcn_s_wait_asynccnt)
  __builtin_amdgcn_s_wait_asynccnt(0);
#else
  asm volatile("s_wait_asynccnt 0x0" ::: "memory");
#endif
#endif
}

// ---------------------------------------------------------------------------
// Kernel 1: convert x fp32 -> f16 into workspace (1 MB, stays L2-resident).
// ---------------------------------------------------------------------------
__global__ __launch_bounds__(256) void cvt_x_kernel(const float* __restrict__ x,
                                                    _Float16* __restrict__ xh) {
  int i = (blockIdx.x * 256 + threadIdx.x) * 4;
  v4f xv = *(const v4f*)(x + i);
  v4h h;
  h[0] = (_Float16)xv[0];
  h[1] = (_Float16)xv[1];
  h[2] = (_Float16)xv[2];
  h[3] = (_Float16)xv[3];
  *(v4h*)(xh + i) = h;
}

// ---------------------------------------------------------------------------
// Kernel 2: WMMA GEMM.
//   block = 128 threads = 4 waves; wave w owns N rows [blk*64 + w*16, +16).
//   Double-buffered LDS chunks of x[0:128, kc:kc+64] (f16, 16KB); each chunk
//   feeds 2 k-substeps of 8 WMMAs per wave. W int32 -> f16 inline.
// ---------------------------------------------------------------------------
#define KC     64   // K-chunk per LDS stage
#define LDS_RS 72   // padded row stride in halfs (144B: 16B-aligned, 36-bank stride)

__global__ __launch_bounds__(128) void gemm_kernel(const _Float16* __restrict__ xh,
                                                   const int* __restrict__ w,
                                                   const float* __restrict__ bias,
                                                   float* __restrict__ out) {
  __shared__ _Float16 atile[2][MTOK * LDS_RS];

  const int tid   = threadIdx.x;
  const int wave  = tid >> 5;
  const int lane  = tid & 31;
  const int lhalf = lane >> 4;  // which half-wave (selects K sub-block)
  const int lrow  = lane & 15;  // M row within A-tile / N col within B-tile
  const int n     = blockIdx.x * 64 + wave * 16 + lrow;

  v8f acc[8] = {};

  // Cooperative copy of x[0:128, k0:k0+KC] (f16) into LDS buffer `buf`.
  // 128 rows * 64 halfs = 1024 chunks of 8 halfs (16B); 8 per thread.
  auto stage = [&](int buf, int k0) {
#pragma unroll
    for (int i = 0; i < 8; ++i) {
      int ch = i * 128 + tid;
      int r  = ch >> 3;             // token row 0..127
      int kk = (ch & 7) << 3;       // 0,8,...,56
#if HAS_ASYNC_LDS
      __builtin_amdgcn_global_load_async_to_lds_b128(
          (__attribute__((address_space(1))) b128i*)(xh + (size_t)r * KCOLS + k0 + kk),
          (__attribute__((address_space(3))) b128i*)&atile[buf][r * LDS_RS + kk],
          0, 0);
#else
      *(v8h*)&atile[buf][r * LDS_RS + kk] =
          *(const v8h*)(xh + (size_t)r * KCOLS + k0 + kk);
#endif
    }
  };

  stage(0, 0);
  wait_async_zero();
  __syncthreads();

  const int* __restrict__ wrow = w + (size_t)n * KCOLS;
  const int NC = KCOLS / KC;  // 64 staged chunks

  for (int c = 0; c < NC; ++c) {
    const int buf = c & 1;
    if (c + 1 < NC) stage(buf ^ 1, (c + 1) * KC);  // async prefetch next chunk

#pragma unroll
    for (int sub = 0; sub < 2; ++sub) {
      const int k0 = c * KC + sub * 32;

      // B fragment: lane holds W[n][k] for k = k0 + 8*lhalf + {0..7} (elems
      // 0..7) and k = k0 + 16 + 8*lhalf + {0..7} (elems 8..15). Exact in f16.
      const int kb = k0 + 8 * lhalf;
      v4i wi0 = *(const v4i*)(wrow + kb);
      v4i wi1 = *(const v4i*)(wrow + kb + 4);
      v4i wi2 = *(const v4i*)(wrow + kb + 16);
      v4i wi3 = *(const v4i*)(wrow + kb + 20);
      v16h bfrag;
#pragma unroll
      for (int i = 0; i < 4; ++i) {
        bfrag[i]      = (_Float16)(float)wi0[i];
        bfrag[i + 4]  = (_Float16)(float)wi1[i];
        bfrag[i + 8]  = (_Float16)(float)wi2[i];
        bfrag[i + 12] = (_Float16)(float)wi3[i];
      }

#pragma unroll
      for (int m = 0; m < 8; ++m) {
        const _Float16* ar =
            &atile[buf][(m * 16 + lrow) * LDS_RS + sub * 32 + 8 * lhalf];
        v8h a0 = *(const v8h*)(ar);       // K = k0 + 8*lhalf + 0..7
        v8h a1 = *(const v8h*)(ar + 16);  // K = k0 + 16 + 8*lhalf + 0..7
        v16h afrag;
#pragma unroll
        for (int i = 0; i < 8; ++i) {
          afrag[i]     = a0[i];
          afrag[i + 8] = a1[i];
        }
        acc[m] = __builtin_amdgcn_wmma_f32_16x16x32_f16(
            false, afrag, false, bfrag, (short)0, acc[m], false, false);
      }
    }

    wait_async_zero();
    __syncthreads();
  }

  // Epilogue: D[v] at lane = out[M = m*16 + v + 8*lhalf][N = n].
  const float bv = bias[n];
#pragma unroll
  for (int m = 0; m < 8; ++m) {
    const int trow = m * 16 + 8 * lhalf;
#pragma unroll
    for (int v = 0; v < 8; ++v) {
      out[(size_t)(trow + v) * NROWS + n] = acc[m][v] * WSCALE + bv;
    }
  }
}

// ---------------------------------------------------------------------------
extern "C" void kernel_launch(void* const* d_in, const int* in_sizes, int n_in,
                              void* d_out, int out_size, void* d_ws, size_t ws_size,
                              hipStream_t stream) {
  const float* x    = (const float*)d_in[0];   // [4,32,4096] fp32
  const int*   wq   = (const int*)d_in[1];     // [11008,4096] int32 (int8 values)
  const float* bias = (const float*)d_in[2];   // [11008] fp32
  float*       out  = (float*)d_out;           // [4,32,11008] fp32
  _Float16*    xh   = (_Float16*)d_ws;         // [128,4096] f16 staging (1 MB)

  // x fp32 -> f16
  cvt_x_kernel<<<(MTOK * KCOLS) / (256 * 4), 256, 0, stream>>>(x, xh);

  // GEMM: 172 blocks x 128 threads; each block computes 128x64 outputs.
  gemm_kernel<<<NROWS / 64, 128, 0, stream>>>(xh, wq, bias, out);
}
